// SeqDecoder_49941879718577
// MI455X (gfx1250) — compile-verified, matrix operand order checked by reading, not run on hardware
//
#include <hip/hip_runtime.h>
#include <hip/hip_bf16.h>

typedef __attribute__((ext_vector_type(16))) __bf16 v16bf;
typedef __attribute__((ext_vector_type(8)))  __bf16 v8bf;
typedef __attribute__((ext_vector_type(8)))  float  v8f;
typedef unsigned short u16;

#define B_  64
#define H_  1024
#define V_  16384
#define T_  64

// ---------- helpers ----------

__device__ __forceinline__ u16 f2bf(float x) {
    unsigned int u = __float_as_uint(x);
    unsigned int r = u + 0x7FFFu + ((u >> 16) & 1u);   // round-to-nearest-even
    return (u16)(r >> 16);
}

// Load a 16x32 bf16 WMMA A/B fragment from a row-major [16 x K] tile.
// Lane L (wave32): row = L&15, K-halves {kb..kb+7, kb+16..kb+23}, kb = (L>>4)*8.
__device__ __forceinline__ v16bf ld_frag(const u16* tile, int stride) {
    const int lane = threadIdx.x & 31;
    const int row  = lane & 15;
    const int kb   = (lane >> 4) << 3;
    const __bf16* p = reinterpret_cast<const __bf16*>(tile) + (size_t)row * stride + kb;
    v16bf f;
    reinterpret_cast<v8bf*>(&f)[0] = *reinterpret_cast<const v8bf*>(p);
    reinterpret_cast<v8bf*>(&f)[1] = *reinterpret_cast<const v8bf*>(p + 16);
    return f;
}

// One K-chunk worth of fragments for a 64xM x 32xN wave tile.
struct Frags {
    v16bf a[4];   // 4 M-tiles of A
    v16bf b[2];   // 2 N-tiles of B
};

__device__ __forceinline__ void ld_chunk(Frags& fr,
                                         const u16* abase, int astride,
                                         const u16* bbase, int bstride,
                                         int kk) {
    fr.b[0] = ld_frag(bbase + kk, bstride);
    fr.b[1] = ld_frag(bbase + (size_t)16 * bstride + kk, bstride);
    #pragma unroll
    for (int mt = 0; mt < 4; ++mt)
        fr.a[mt] = ld_frag(abase + (size_t)(mt * 16) * astride + kk, astride);
}

__device__ __forceinline__ void mma_chunk(v8f acc[4][2], const Frags& fr) {
    #pragma unroll
    for (int mt = 0; mt < 4; ++mt) {
        acc[mt][0] = __builtin_amdgcn_wmma_f32_16x16x32_bf16(
            false, fr.a[mt], false, fr.b[0], (short)0, acc[mt][0], false, false);
        acc[mt][1] = __builtin_amdgcn_wmma_f32_16x16x32_bf16(
            false, fr.a[mt], false, fr.b[1], (short)0, acc[mt][1], false, false);
    }
}

// Double-buffered 64x32 GEMM tile over K: prefetch chunk kk+32 while MMA'ing kk.
__device__ __forceinline__ void gemm_tile(v8f acc[4][2],
                                          const u16* abase, int astride,
                                          const u16* bbase, int bstride,
                                          int K) {
    Frags fr[2];
    ld_chunk(fr[0], abase, astride, bbase, bstride, 0);
    for (int kk = 0; kk < K - 32; kk += 32) {
        const int cu = (kk >> 5) & 1;
        ld_chunk(fr[cu ^ 1], abase, astride, bbase, bstride, kk + 32);
        mma_chunk(acc, fr[cu]);
    }
    mma_chunk(acc, fr[((K - 32) >> 5) & 1]);
}

// ---------- init / conversion kernels ----------

__global__ void zero_predicts_kernel(float4* p, long n4) {
    long i = (long)blockIdx.x * blockDim.x + threadIdx.x;
    long stride = (long)gridDim.x * blockDim.x;
    float4 z = make_float4(0.f, 0.f, 0.f, 0.f);
    for (; i < n4; i += stride) p[i] = z;
}

// Wg[r, 0:H] = W_ih[r,:], Wg[r, H:2H] = W_hh[r,:]   (4H x 2H bf16)
__global__ void cvt_gate_weights_kernel(const float* __restrict__ wih,
                                        const float* __restrict__ whh,
                                        u16* __restrict__ wg) {
    long n = (long)(4 * H_) * (2 * H_);
    long i = (long)blockIdx.x * blockDim.x + threadIdx.x;
    long stride = (long)gridDim.x * blockDim.x;
    for (; i < n; i += stride) {
        long r = i >> 11;           // / 2048
        long c = i & 2047;
        float s = (c < H_) ? wih[r * H_ + c] : whh[r * H_ + (c - H_)];
        wg[i] = f2bf(s);
    }
}

__global__ void cvt_out_weights_kernel(const float* __restrict__ wout,
                                       u16* __restrict__ wo) {
    long n = (long)V_ * H_;
    long i = (long)blockIdx.x * blockDim.x + threadIdx.x;
    long stride = (long)gridDim.x * blockDim.x;
    for (; i < n; i += stride) wo[i] = f2bf(wout[i]);
}

__global__ void init_state_kernel(const float* __restrict__ enc_h,
                                  const float* __restrict__ enc_c,
                                  const float* __restrict__ x0,   // [1,H]
                                  u16* __restrict__ xh0,          // [B,2H]
                                  float* __restrict__ c,          // [B,H]
                                  float* __restrict__ mask) {     // [B]
    int i = blockIdx.x * blockDim.x + threadIdx.x;   // covers B*2H = 131072
    if (i < B_ * 2 * H_) {
        int b = i >> 11;
        int k = i & 2047;
        float s = (k < H_) ? x0[k] : enc_h[b * H_ + (k - H_)];
        xh0[i] = f2bf(s);
    }
    if (i < B_ * H_) c[i] = enc_c[i];
    if (i < B_)      mask[i] = 1.0f;
}

// ---------- step kernels ----------

// Gates GEMM + LSTM pointwise. Grid: H/32 blocks x 128 threads (4 waves).
// Wave w computes gate segment w (i/f/g/o) for 32 hidden units of this block:
// a 64x32 f32 tile via WMMA over K=2048 (x||h fused).
__global__ __launch_bounds__(128) void gates_step_kernel(
        const u16* __restrict__ xh_cur,   // [B, 2H] bf16
        u16* __restrict__ xh_next,        // h written into [:, H:2H]
        const u16* __restrict__ wg,       // [4H, 2H] bf16
        const float* __restrict__ b_ih,
        const float* __restrict__ b_hh,
        float* __restrict__ c) {          // [B, H]
    __shared__ float lds[4][B_][32];
    const int w    = threadIdx.x >> 5;    // wave = gate segment
    const int lane = threadIdx.x & 31;
    const int n0   = blockIdx.x * 32;

    v8f acc[4][2] = {};
    const u16* brow = wg + (size_t)(w * H_ + n0) * (2 * H_);

    gemm_tile(acc, xh_cur, 2 * H_, brow, 2 * H_, 2 * H_);

    // spill C tiles: lane holds col N=lane&15, rows r + 8*(lane>=16)
    const int cn = lane & 15, hi = lane >> 4;
    #pragma unroll
    for (int mt = 0; mt < 4; ++mt)
        #pragma unroll
        for (int nt = 0; nt < 2; ++nt)
            #pragma unroll
            for (int r = 0; r < 8; ++r)
                lds[w][mt * 16 + hi * 8 + r][nt * 16 + cn] = acc[mt][nt][r];
    __syncthreads();

    // pointwise LSTM update for columns [n0, n0+32)
    for (int e = threadIdx.x; e < B_ * 32; e += 128) {
        int m = e >> 5, n = e & 31;
        int col = n0 + n;
        float gi = lds[0][m][n] + b_ih[0 * H_ + col] + b_hh[0 * H_ + col];
        float gf = lds[1][m][n] + b_ih[1 * H_ + col] + b_hh[1 * H_ + col];
        float gg = lds[2][m][n] + b_ih[2 * H_ + col] + b_hh[2 * H_ + col];
        float go = lds[3][m][n] + b_ih[3 * H_ + col] + b_hh[3 * H_ + col];
        float i_ = 1.f / (1.f + __expf(-gi));
        float f_ = 1.f / (1.f + __expf(-gf));
        float g_ = tanhf(gg);
        float o_ = 1.f / (1.f + __expf(-go));
        float cn_ = f_ * c[m * H_ + col] + i_ * g_;
        c[m * H_ + col] = cn_;
        float hn = o_ * tanhf(cn_);
        xh_next[(size_t)m * (2 * H_) + H_ + col] = f2bf(hn);
    }
}

// Logits GEMM + coalesced store + per-block partial argmax.
// Grid: V/128 blocks x 128 threads; block owns a 64x128 logits tile.
__global__ __launch_bounds__(128) void logits_step_kernel(
        const u16* __restrict__ xh_next,  // h at [:, H:2H]
        const u16* __restrict__ wo,       // [V, H] bf16
        const float* __restrict__ b_out,
        float* __restrict__ out_logits,   // [B, V] slice for step t
        float* __restrict__ pval,         // [B, gridDim]
        int*   __restrict__ pidx) {
    __shared__ float lds[B_][128];
    const int w    = threadIdx.x >> 5;
    const int lane = threadIdx.x & 31;
    const int v0   = blockIdx.x * 128;
    const int vb   = v0 + w * 32;

    v8f acc[4][2] = {};
    const u16* arow = xh_next + H_;                 // row stride 2H
    const u16* brow = wo + (size_t)vb * H_;

    gemm_tile(acc, arow, 2 * H_, brow, H_, H_);

    const int cn = lane & 15, hi = lane >> 4;
    #pragma unroll
    for (int mt = 0; mt < 4; ++mt)
        #pragma unroll
        for (int nt = 0; nt < 2; ++nt) {
            float bias = b_out[vb + nt * 16 + cn];
            #pragma unroll
            for (int r = 0; r < 8; ++r)
                lds[mt * 16 + hi * 8 + r][w * 32 + nt * 16 + cn] = acc[mt][nt][r] + bias;
        }
    __syncthreads();

    // coalesced b128 stores of the tile
    const float4* l4 = reinterpret_cast<const float4*>(&lds[0][0]);
    for (int i = threadIdx.x; i < B_ * 32; i += 128) {
        int row = i >> 5, c4 = i & 31;
        *reinterpret_cast<float4*>(&out_logits[(size_t)row * V_ + v0 + c4 * 4]) = l4[i];
    }

    // partial argmax per row over this block's 128 columns (first-max wins)
    if (threadIdx.x < B_) {
        int m = threadIdx.x;
        float best = lds[m][0]; int bi = 0;
        for (int j = 1; j < 128; ++j) {
            float v = lds[m][j];
            if (v > best) { best = v; bi = j; }
        }
        pval[m * (V_ / 128) + blockIdx.x] = best;
        pidx[m * (V_ / 128) + blockIdx.x] = v0 + bi;
    }
}

// Final argmax reduce, predict/mask write, embedding gather -> bf16 x_next.
__global__ __launch_bounds__(128) void finalize_step_kernel(
        const float* __restrict__ pval, const int* __restrict__ pidx,
        float* __restrict__ predicts_t,   // [B, V] slice (pre-zeroed)
        float* __restrict__ masks_t,      // [B] slice
        float* __restrict__ mask,         // [B] running mask
        u16*   __restrict__ xh_next,      // x written into [:, 0:H]
        const float* __restrict__ emb,    // [V, H]
        const int* __restrict__ eos_p) {
    __shared__ int sel[B_];
    const int nb = V_ / 128;
    int tid = threadIdx.x;
    if (tid < B_) {
        float best = pval[tid * nb]; int bi = pidx[tid * nb];
        for (int j = 1; j < nb; ++j) {
            float v = pval[tid * nb + j];
            int  ix = pidx[tid * nb + j];
            if (v > best || (v == best && ix < bi)) { best = v; bi = ix; }
        }
        sel[tid] = bi;
        predicts_t[(size_t)tid * V_ + bi] = 1.0f;
        float mold = mask[tid];
        masks_t[tid] = mold;                          // masks output = pre-update mask
        mask[tid] = (bi == eos_p[0]) ? 0.0f : mold;
    }
    __syncthreads();
    for (int e = tid; e < B_ * H_; e += 128) {
        int b = e >> 10, k = e & (H_ - 1);
        xh_next[(size_t)b * (2 * H_) + k] = f2bf(emb[(size_t)sel[b] * H_ + k]);
    }
}

// ---------- host launcher ----------

extern "C" void kernel_launch(void* const* d_in, const int* in_sizes, int n_in,
                              void* d_out, int out_size, void* d_ws, size_t ws_size,
                              hipStream_t stream) {
    const float* enc_h = (const float*)d_in[0];
    const float* enc_c = (const float*)d_in[1];
    const float* wih   = (const float*)d_in[2];
    const float* whh   = (const float*)d_in[3];
    const float* bih   = (const float*)d_in[4];
    const float* bhh   = (const float*)d_in[5];
    const float* wout  = (const float*)d_in[6];
    const float* bout  = (const float*)d_in[7];
    const float* emb   = (const float*)d_in[8];
    const float* x0    = (const float*)d_in[9];
    const int*   eos   = (const int*)d_in[11];

    float* out      = (float*)d_out;
    float* predicts = out;                                  // [T,B,V]
    float* logits   = out + (size_t)T_ * B_ * V_;           // [T,B,V]
    float* masks    = out + 2 * (size_t)T_ * B_ * V_;       // [T,1,B]

    // workspace layout (all offsets 256B-aligned)
    char* ws = (char*)d_ws;
    u16*   wg    = (u16*)(ws);                               // 4H*2H bf16 = 16 MB
    u16*   wo    = (u16*)(ws + 16777216);                    // V*H bf16   = 32 MB
    u16*   xh    = (u16*)(ws + 16777216 + 33554432);         // 2*B*2H bf16
    float* cbuf  = (float*)(ws + 16777216 + 33554432 + 524288);
    float* mbuf  = (float*)(ws + 16777216 + 33554432 + 524288 + 262144);
    float* pvalb = (float*)(ws + 16777216 + 33554432 + 524288 + 262144 + 256);
    int*   pidxb = (int*)  (ws + 16777216 + 33554432 + 524288 + 262144 + 256 + 32768);

    // one-time (per call) init: zero predicts, bf16 weights, state
    zero_predicts_kernel<<<4096, 256, 0, stream>>>(
        (float4*)predicts, (long)T_ * B_ * V_ / 4);
    cvt_gate_weights_kernel<<<4096, 256, 0, stream>>>(wih, whh, wg);
    cvt_out_weights_kernel<<<8192, 256, 0, stream>>>(wout, wo);
    init_state_kernel<<<(B_ * 2 * H_ + 255) / 256, 256, 0, stream>>>(
        enc_h, enc_c, x0, xh, cbuf, mbuf);

    for (int t = 0; t < T_; ++t) {
        u16* cur = xh + (size_t)(t & 1)       * B_ * 2 * H_;
        u16* nxt = xh + (size_t)((t + 1) & 1) * B_ * 2 * H_;

        gates_step_kernel<<<H_ / 32, 128, 0, stream>>>(
            cur, nxt, wg, bih, bhh, cbuf);
        logits_step_kernel<<<V_ / 128, 128, 0, stream>>>(
            nxt, wo, bout, logits + (size_t)t * B_ * V_, pvalb, pidxb);
        finalize_step_kernel<<<1, 128, 0, stream>>>(
            pvalb, pidxb, predicts + (size_t)t * B_ * V_, masks + (size_t)t * B_,
            mbuf, nxt, emb, eos);
    }
}
